// CompactS6Layer_6313601925546
// MI455X (gfx1250) — matrix operand order repaired and checked by reading, not compiled, not run on hardware
//
#include <hip/hip_runtime.h>

#define D_MODEL 1024
#define D_STATE 8
#define BATCH   8
#define SEQLEN  2048
#define BL      (BATCH * SEQLEN)   // 16384 rows

typedef __attribute__((ext_vector_type(16))) __bf16 v16bf;
typedef __attribute__((ext_vector_type(8)))  float  v8f;
typedef int v4i_vs __attribute__((vector_size(16)));   // matches async-LDS builtin param

union FragAB { v16bf v; unsigned u[8]; };

#if defined(__has_builtin)
#if __has_builtin(__builtin_amdgcn_global_load_async_to_lds_b128) && \
    __has_builtin(__builtin_amdgcn_s_wait_asynccnt)
#define HAVE_ASYNC_LDS 1
#endif
#endif
#ifndef HAVE_ASYNC_LDS
#define HAVE_ASYNC_LDS 0
#endif

__device__ __forceinline__ unsigned short f2bf_bits(float f) {
  union { float f; unsigned u; } v; v.f = f;
  unsigned r = (v.u + 0x7FFFu + ((v.u >> 16) & 1u)) >> 16;   // RNE
  return (unsigned short)r;
}
__device__ __forceinline__ float bf2f(unsigned short s) {
  union { unsigned u; float f; } v; v.u = ((unsigned)s) << 16;
  return v.f;
}

// ---------------------------------------------------------------------------
// Tiled bf16 WMMA GEMM: C[M,N] = A[M,K] * B[K,N] with B pre-transposed to
// BT[N][K] bf16. Block tile 128x128, BK=32 (== WMMA K). 256 threads = 8 waves
// in 2(M)x4(N); each wave owns 64x32 = 8 accumulators of 16x16.
// Double-buffered LDS; tiles stream in via GLOBAL_LOAD_ASYNC_TO_LDS_B128
// (ASYNCcnt) when available, overlapping the WMMA stream of the live buffer.
// Out-of-range B rows are index-clamped (their outputs are never stored), so
// the staging path is branch-free.
// MODE 0: split epilogue -> bf16 x_in / bf16 z
// MODE 1: +bias, softplus -> fp32 dt
// MODE 2: plain fp32 store (B/C projection, N=16)
// MODE 3: residual add (out = x + A*B) -> fp32
// ---------------------------------------------------------------------------
template <int MODE>
__global__ __launch_bounds__(256) void wmma_gemm(
    const unsigned short* __restrict__ A,    // [M][K] bf16
    const unsigned short* __restrict__ BT,   // [N][K] bf16 (pre-transposed)
    int M, int K, int N,
    float* __restrict__ out0,
    unsigned short* __restrict__ bf0,
    unsigned short* __restrict__ bf1,
    const float* __restrict__ bias,
    const float* __restrict__ addsrc)
{
  __shared__ unsigned short As[2][128][40];  // +pad: 80B row stride, 16B aligned
  __shared__ unsigned short Bs[2][128][40];  // N-major (Bs[n][k])

  const int tid  = threadIdx.x;
  const int lane = tid & 31;
  const int wave = tid >> 5;
  const int waveM = wave & 1;               // 2 waves in M, 64 rows each
  const int waveN = wave >> 1;              // 4 waves in N, 32 cols each
  const int blockRow0 = blockIdx.x * 128;
  const int blockCol0 = blockIdx.y * 128;

  v8f acc[4][2] = {};

  const int ldrow = tid >> 1;               // 0..127
  const int ldk   = (tid & 1) * 16;         // 0 or 16

  const unsigned short* Aptr = A + (size_t)(blockRow0 + ldrow) * K + ldk;
  const int bn  = blockCol0 + ldrow;
  const int bnc = (bn < N) ? bn : (N - 1);  // clamp: padded cols never stored
  const unsigned short* Bptr = BT + (size_t)bnc * K + ldk;

  auto stage = [&](int k0, int buf) {
#if HAVE_ASYNC_LDS
    __builtin_amdgcn_global_load_async_to_lds_b128(
        (v4i_vs*)(Aptr + k0), (v4i_vs*)&As[buf][ldrow][ldk], 0, 0);
    __builtin_amdgcn_global_load_async_to_lds_b128(
        (v4i_vs*)(Aptr + k0), (v4i_vs*)&As[buf][ldrow][ldk], 16, 0);
    __builtin_amdgcn_global_load_async_to_lds_b128(
        (v4i_vs*)(Bptr + k0), (v4i_vs*)&Bs[buf][ldrow][ldk], 0, 0);
    __builtin_amdgcn_global_load_async_to_lds_b128(
        (v4i_vs*)(Bptr + k0), (v4i_vs*)&Bs[buf][ldrow][ldk], 16, 0);
#else
    uint4 a0g = *(const uint4*)(Aptr + k0);
    uint4 a1g = *(const uint4*)(Aptr + k0 + 8);
    uint4 b0g = *(const uint4*)(Bptr + k0);
    uint4 b1g = *(const uint4*)(Bptr + k0 + 8);
    *(uint4*)&As[buf][ldrow][ldk]     = a0g;
    *(uint4*)&As[buf][ldrow][ldk + 8] = a1g;
    *(uint4*)&Bs[buf][ldrow][ldk]     = b0g;
    *(uint4*)&Bs[buf][ldrow][ldk + 8] = b1g;
    if (k0 + 32 < K) {                       // gfx1250 global_prefetch_b8
      __builtin_prefetch(Aptr + k0 + 32, 0, 3);
      __builtin_prefetch(Bptr + k0 + 32, 0, 3);
    }
#endif
  };

  const int ntiles = K >> 5;                 // K / 32
  stage(0, 0);

  for (int i = 0; i < ntiles; ++i) {
    const int cur = i & 1;
#if HAVE_ASYNC_LDS
    __builtin_amdgcn_s_wait_asynccnt(0);     // my async loads into buf[cur] done
#endif
    __syncthreads();                          // everyone's loads done / reads drained
    if (i + 1 < ntiles) stage((i + 1) << 5, cur ^ 1);  // overlap next tile

    // ---- fragments per documented 16-bit WMMA VGPR layouts ----
    FragAB afr[4], bfr[2];
    const int ar = lane & 15;
    const int kb = (lane >> 4) * 8;          // A: lanes 0-15 K 0..7/16..23, 16-31 K 8..15/24..31
#pragma unroll
    for (int mi = 0; mi < 4; ++mi) {
      const int r = waveM * 64 + mi * 16 + ar;
      *(uint4*)&afr[mi].u[0] = *(const uint4*)&As[cur][r][kb];
      *(uint4*)&afr[mi].u[4] = *(const uint4*)&As[cur][r][16 + kb];
    }
    const int kh = (lane >> 4) * 8;          // B: lanes 0-15 K 0..7/16..23, 16-31 K 8..15/24..31
#pragma unroll
    for (int ni = 0; ni < 2; ++ni) {
      const int n = waveN * 32 + ni * 16 + (lane & 15);
      *(uint4*)&bfr[ni].u[0] = *(const uint4*)&Bs[cur][n][kh];
      *(uint4*)&bfr[ni].u[4] = *(const uint4*)&Bs[cur][n][kh + 16];
    }
#pragma unroll
    for (int mi = 0; mi < 4; ++mi)
#pragma unroll
      for (int ni = 0; ni < 2; ++ni)
        acc[mi][ni] = __builtin_amdgcn_wmma_f32_16x16x32_bf16(
            false, afr[mi].v, false, bfr[ni].v, (short)0, acc[mi][ni], false, false);
    __syncthreads();                          // reads of buf[cur] drained
  }

  // ---- epilogue: 16x16 f32 C/D layout (VGPR r: M = r + 8*(lane>=16)) ----
#pragma unroll
  for (int mi = 0; mi < 4; ++mi)
#pragma unroll
    for (int ni = 0; ni < 2; ++ni)
#pragma unroll
      for (int r = 0; r < 8; ++r) {
        const int grow = blockRow0 + waveM * 64 + mi * 16 + r + ((lane >> 4) * 8);
        const int gcol = blockCol0 + waveN * 32 + ni * 16 + (lane & 15);
        if (gcol >= N) continue;
        const float val = acc[mi][ni][r];
        if (MODE == 0) {
          if (gcol < D_MODEL) bf0[(size_t)grow * D_MODEL + gcol] = f2bf_bits(val);
          else                bf1[(size_t)grow * D_MODEL + (gcol - D_MODEL)] = f2bf_bits(val);
        } else if (MODE == 1) {
          const float v  = val + bias[gcol];
          const float sp = (v > 20.f) ? v : __logf(1.f + __expf(v));
          out0[(size_t)grow * N + gcol] = sp;
        } else if (MODE == 2) {
          out0[(size_t)grow * N + gcol] = val;
        } else {
          out0[(size_t)grow * N + gcol] = addsrc[(size_t)grow * N + gcol] + val;
        }
      }
}

// ---------------------------------------------------------------------------
// Tiled transpose + fp32->bf16 cast: out[C][R] = bf16(in[R][C]).
// 32x32 tiles via LDS; one-time weight staging cost.
// ---------------------------------------------------------------------------
__global__ __launch_bounds__(256) void transpose_cast_kernel(
    const float* __restrict__ in, unsigned short* __restrict__ out, int R, int C)
{
  __shared__ float tile[32][33];
  const int r0 = blockIdx.x * 32;
  const int c0 = blockIdx.y * 32;
  const int tx = threadIdx.x & 31;
  const int ty = threadIdx.x >> 5;          // 0..7
#pragma unroll
  for (int p = 0; p < 4; ++p)
    tile[ty + p * 8][tx] = in[(size_t)(r0 + ty + p * 8) * C + (c0 + tx)];
  __syncthreads();
#pragma unroll
  for (int p = 0; p < 4; ++p)
    out[(size_t)(c0 + ty + p * 8) * R + (r0 + tx)] = f2bf_bits(tile[tx][ty + p * 8]);
}

// W_B|W_C [D,8]+[D,8] -> transposed packed [16][D] bf16
__global__ void pack_wbc_t_kernel(const float* __restrict__ WB,
                                  const float* __restrict__ WC,
                                  unsigned short* __restrict__ out) {
  const int i = blockIdx.x * 256 + threadIdx.x;   // i = k*8 + n
  if (i < D_MODEL * D_STATE) {
    const int k = i >> 3, n = i & 7;
    out[(size_t)n * D_MODEL + k]       = f2bf_bits(WB[i]);
    out[(size_t)(8 + n) * D_MODEL + k] = f2bf_bits(WC[i]);
  }
}

// ---------------------------------------------------------------------------
// LayerNorm over D=1024, one row per 256-thread block, bf16 output.
// ---------------------------------------------------------------------------
__global__ __launch_bounds__(256) void layernorm_kernel(
    const float* __restrict__ x, const float* __restrict__ w,
    const float* __restrict__ b, unsigned short* __restrict__ out)
{
  __shared__ float red[256];
  const int row = blockIdx.x;
  const int tid = threadIdx.x;
  const float* xr = x + (size_t)row * D_MODEL;
  float s = 0.f, s2 = 0.f;
#pragma unroll
  for (int i = tid; i < D_MODEL; i += 256) { float v = xr[i]; s += v; s2 += v * v; }
  red[tid] = s; __syncthreads();
  for (int o = 128; o > 0; o >>= 1) { if (tid < o) red[tid] += red[tid + o]; __syncthreads(); }
  const float mu = red[0] * (1.f / D_MODEL); __syncthreads();
  red[tid] = s2; __syncthreads();
  for (int o = 128; o > 0; o >>= 1) { if (tid < o) red[tid] += red[tid + o]; __syncthreads(); }
  const float var  = red[0] * (1.f / D_MODEL) - mu * mu;
  const float rstd = rsqrtf(var + 1e-5f);
#pragma unroll
  for (int i = tid; i < D_MODEL; i += 256)
    out[(size_t)row * D_MODEL + i] = f2bf_bits((xr[i] - mu) * rstd * w[i] + b[i]);
}

// ---------------------------------------------------------------------------
// Sequential selective scan: one thread per (b,d), N=8 fp32 state in regs.
// ---------------------------------------------------------------------------
__global__ __launch_bounds__(256) void scan_kernel(
    const float* __restrict__ dt, const unsigned short* __restrict__ xin,
    const float* __restrict__ BC, const float* __restrict__ A_log,
    const float* __restrict__ h_prev, float* __restrict__ y,
    float* __restrict__ h_final)
{
  const int d = blockIdx.x * 256 + threadIdx.x;   // 0..1023
  const int b = blockIdx.y;                       // 0..7
  float a[D_STATE], h[D_STATE];
#pragma unroll
  for (int n = 0; n < D_STATE; ++n) {
    a[n] = -__expf(A_log[d * D_STATE + n]);
    h[n] = h_prev[((size_t)b * D_MODEL + d) * D_STATE + n];
  }
  for (int t = 0; t < SEQLEN; ++t) {
    const size_t rb  = (size_t)b * SEQLEN + t;
    const float dtv = dt[rb * D_MODEL + d];
    const float xv  = bf2f(xin[rb * D_MODEL + d]);
    const float dx  = dtv * xv;
    const float* bc = BC + rb * 16;               // cols 0..7 = B_t, 8..15 = C_t
    float yv = 0.f;
#pragma unroll
    for (int n = 0; n < D_STATE; ++n) {
      const float dA = __expf(dtv * a[n]);
      h[n] = dA * h[n] + dx * bc[n];
      yv  += h[n] * bc[8 + n];
    }
    y[rb * D_MODEL + d] = yv;
  }
#pragma unroll
  for (int n = 0; n < D_STATE; ++n)
    h_final[((size_t)b * D_MODEL + d) * D_STATE + n] = h[n];
}

// ---------------------------------------------------------------------------
// ssm = y * silu(z) + x_in * D_skip  ->  bf16
// ---------------------------------------------------------------------------
__global__ __launch_bounds__(256) void gate_kernel(
    const float* __restrict__ y, const unsigned short* __restrict__ z,
    const unsigned short* __restrict__ xin, const float* __restrict__ Dskip,
    unsigned short* __restrict__ ssm)
{
  const size_t i = (size_t)blockIdx.x * 256 + threadIdx.x;
  const int col = (int)(i & (D_MODEL - 1));
  const float zv  = bf2f(z[i]);
  const float sil = zv / (1.f + __expf(-zv));
  const float v   = y[i] * sil + bf2f(xin[i]) * Dskip[col];
  ssm[i] = f2bf_bits(v);
}

extern "C" void kernel_launch(void* const* d_in, const int* in_sizes, int n_in,
                              void* d_out, int out_size, void* d_ws, size_t ws_size,
                              hipStream_t stream) {
  (void)in_sizes; (void)n_in; (void)out_size; (void)ws_size;
  const float* x      = (const float*)d_in[0];
  const float* h_prev = (const float*)d_in[1];
  const float* norm_w = (const float*)d_in[2];
  const float* norm_b = (const float*)d_in[3];
  const float* W_in_g = (const float*)d_in[4];
  const float* W_dt   = (const float*)d_in[5];
  const float* b_dt   = (const float*)d_in[6];
  const float* A_log  = (const float*)d_in[7];
  const float* W_B    = (const float*)d_in[8];
  const float* W_C    = (const float*)d_in[9];
  const float* D_skip = (const float*)d_in[10];
  const float* W_out  = (const float*)d_in[11];

  float* out     = (float*)d_out;
  float* h_final = out + (size_t)BATCH * SEQLEN * D_MODEL;

  char* ws = (char*)d_ws;
  size_t off = 0;
  auto alloc = [&](size_t bytes) -> void* {
    void* p = ws + off;
    off += (bytes + 255) & ~(size_t)255;
    return p;
  };
  unsigned short* xnorm_bf = (unsigned short*)alloc((size_t)BL * D_MODEL * 2);
  unsigned short* WgT_bf   = (unsigned short*)alloc((size_t)2 * D_MODEL * D_MODEL * 2);  // [2048][1024]
  unsigned short* WdtT_bf  = (unsigned short*)alloc((size_t)D_MODEL * D_MODEL * 2);      // [1024][1024]
  unsigned short* WoutT_bf = (unsigned short*)alloc((size_t)D_MODEL * D_MODEL * 2);      // [1024][1024]
  unsigned short* WbcT_bf  = (unsigned short*)alloc((size_t)16 * D_MODEL * 2);           // [16][1024]
  unsigned short* xin_bf   = (unsigned short*)alloc((size_t)BL * D_MODEL * 2);
  unsigned short* z_bf     = (unsigned short*)alloc((size_t)BL * D_MODEL * 2);
  float*          dt_f     = (float*)alloc((size_t)BL * D_MODEL * 4);
  float*          BC_f     = (float*)alloc((size_t)BL * 16 * 4);
  float*          y_f      = (float*)alloc((size_t)BL * D_MODEL * 4);
  unsigned short* ssm_bf   = xnorm_bf;   // xnorm dead after GEMM1 -> reuse

  // One-time weight staging: transpose + cast to bf16 [N][K]
  {
    dim3 gt(D_MODEL / 32, (2 * D_MODEL) / 32);
    transpose_cast_kernel<<<gt, 256, 0, stream>>>(W_in_g, WgT_bf, D_MODEL, 2 * D_MODEL);
    dim3 gq(D_MODEL / 32, D_MODEL / 32);
    transpose_cast_kernel<<<gq, 256, 0, stream>>>(W_dt, WdtT_bf, D_MODEL, D_MODEL);
    transpose_cast_kernel<<<gq, 256, 0, stream>>>(W_out, WoutT_bf, D_MODEL, D_MODEL);
    pack_wbc_t_kernel<<<(D_MODEL * D_STATE + 255) / 256, 256, 0, stream>>>(W_B, W_C, WbcT_bf);
  }

  // LayerNorm -> bf16
  layernorm_kernel<<<BL, 256, 0, stream>>>(x, norm_w, norm_b, xnorm_bf);

  // GEMM1: x_norm @ W_in_gate -> split x_in / z (bf16)
  dim3 g1(BL / 128, (2 * D_MODEL) / 128);
  wmma_gemm<0><<<g1, 256, 0, stream>>>(xnorm_bf, WgT_bf, BL, D_MODEL, 2 * D_MODEL,
                                       nullptr, xin_bf, z_bf, nullptr, nullptr);

  // GEMM2: x_in @ W_dt + b_dt, softplus -> fp32 dt
  dim3 g2(BL / 128, D_MODEL / 128);
  wmma_gemm<1><<<g2, 256, 0, stream>>>(xin_bf, WdtT_bf, BL, D_MODEL, D_MODEL,
                                       dt_f, nullptr, nullptr, b_dt, nullptr);

  // GEMM3: x_in @ [W_B | W_C] -> fp32 [BL,16]
  dim3 g3(BL / 128, 1);
  wmma_gemm<2><<<g3, 256, 0, stream>>>(xin_bf, WbcT_bf, BL, D_MODEL, 16,
                                       BC_f, nullptr, nullptr, nullptr, nullptr);

  // Sequential scan over L; h_final is output #2
  dim3 gs(D_MODEL / 256, BATCH);
  scan_kernel<<<gs, 256, 0, stream>>>(dt_f, xin_bf, BC_f, A_log, h_prev, y_f, h_final);

  // Gate + skip -> bf16 ssm
  gate_kernel<<<(BL * D_MODEL) / 256, 256, 0, stream>>>(y_f, z_bf, xin_bf, D_skip, ssm_bf);

  // GEMM4: ssm @ W_out + x -> d_out
  wmma_gemm<3><<<g2, 256, 0, stream>>>(ssm_bf, WoutT_bf, BL, D_MODEL, D_MODEL,
                                       out, nullptr, nullptr, nullptr, x);
}
